// PersonaExpertStack_53412213293219
// MI455X (gfx1250) — matrix operand order repaired and checked by reading, not compiled, not run on hardware
//
#include <hip/hip_runtime.h>

// ---------------------------------------------------------------------------
// PersonaExpertStack: per-layer SwiGLU MLP stack on MI455X (gfx1250, wave32)
//   gate = h·Wg^T ; up = h·Wu^T ; act = silu(gate)*up ; out = act·Wd^T
// Compute-bound (AI = 128 FLOP/byte vs 23.3 TB/s HBM) -> bf16 WMMA path,
// f32 accumulate. Weights read from HBM exactly once (convert-in-kernel);
// hb (16.8 MB) and act (46 MB) intermediates are L2-resident (192 MB L2).
// d_ws layout: [ act bf16 (B,N,I) 46.1 MB | h bf16 (B,N,D) 16.8 MB ]
// Double-buffered LDS weight staging: ONE barrier per K-step.
// ---------------------------------------------------------------------------

typedef __attribute__((ext_vector_type(16))) __bf16 v16bf;
typedef __attribute__((ext_vector_type(8)))  __bf16 v8bf;
typedef __attribute__((ext_vector_type(8)))  float  v8f;

static constexpr int kNL = 32;     // layers
static constexpr int kD  = 1024;   // hidden
static constexpr int kI  = 2816;   // intermediate
static constexpr int kB  = 256;    // batch
static constexpr int SSTR = 40;    // LDS row stride (bf16): conflict-free, 16B aligned
static constexpr int SBUF = 64 * SSTR;                        // one stage buffer (bf16)
static constexpr size_t kActBytes = (size_t)kB * kNL * kI * 2;  // 46,137,344

union BVec { v16bf v; v8bf h[2]; };

__device__ __forceinline__ v8bf cvt8(float4 a, float4 b) {
    v8bf r;
    r[0] = (__bf16)a.x; r[1] = (__bf16)a.y; r[2] = (__bf16)a.z; r[3] = (__bf16)a.w;
    r[4] = (__bf16)b.x; r[5] = (__bf16)b.y; r[6] = (__bf16)b.z; r[7] = (__bf16)b.w;
    return r;
}

// B-operand (32x16 bf16): lane = column, element e -> K = (e/8)*16 + (lane/16)*8 + e%8
__device__ __forceinline__ v16bf lds_b(const __bf16* s) {
    BVec b;
    b.h[0] = *(const v8bf*)(s);
    b.h[1] = *(const v8bf*)(s + 16);
    return b.v;
}

// A-operand (16x32 bf16): lane = row, element e -> K = (e/8)*16 + (lane/16)*8 + e%8
__device__ __forceinline__ v16bf glb_a(const __bf16* p) {
    BVec a;
    a.h[0] = *(const v8bf*)(p);
    a.h[1] = *(const v8bf*)(p + 16);
    return a.v;
}

__device__ __forceinline__ v8f wmma_bf16(v16bf a, v16bf b, v8f c) {
    return __builtin_amdgcn_wmma_f32_16x16x32_bf16(false, a, false, b, (short)0, c,
                                                   false, false);
}

// ---------------------------------------------------------------------------
// Kernel 0: one-shot h (f32) -> bf16 copy in workspace (read once, reuse 44x)
// ---------------------------------------------------------------------------
__global__ __launch_bounds__(256)
void cvt_h_kernel(const float* __restrict__ h, __bf16* __restrict__ hb)
{
    const size_t i = ((size_t)blockIdx.x * 256 + threadIdx.x) * 8;
    *(v8bf*)(hb + i) = cvt8(*(const float4*)(h + i), *(const float4*)(h + i + 4));
}

// ---------------------------------------------------------------------------
// Kernel 1: fused gate/up GEMMs + SiLU*up -> act (bf16 in workspace)
// grid: (kI/64, kNL); block: 256 (8 waves, each owns 32 batch rows)
// ---------------------------------------------------------------------------
__global__ __launch_bounds__(256, 1)
void gateup_silu_kernel(const __bf16* __restrict__ hb,
                        const float* __restrict__ Wg,
                        const float* __restrict__ Wu,
                        __bf16* __restrict__ act)
{
    const int n   = blockIdx.y;
    const int i0  = blockIdx.x * 64;
    const int tid = threadIdx.x;
    const int lane = tid & 31;
    const int wave = tid >> 5;
    const int lrow = lane & 15;   // M row / N col within tile
    const int lhi  = lane >> 4;   // lane half

    __shared__ __bf16 sG[2 * SBUF];
    __shared__ __bf16 sU[2 * SBUF];

    v8f accG[2][4], accU[2][4];
#pragma unroll
    for (int mt = 0; mt < 2; ++mt)
#pragma unroll
        for (int t = 0; t < 4; ++t) { accG[mt][t] = {}; accU[mt][t] = {}; }

    // cooperative weight-tile load: 256 thr -> 64 rows x 4 chunks of 8 f32
    const int wrow = tid >> 2;
    const int wk   = (tid & 3) * 8;
    const float* gRow = Wg + ((size_t)n * kI + (i0 + wrow)) * kD + wk;
    const float* uRow = Wu + ((size_t)n * kI + (i0 + wrow)) * kD + wk;
    const int sOff = wrow * SSTR + wk;

    // per-wave A row base pointers (hb is (B, N, D) bf16)
    const int m0 = wave * 32;
    const __bf16* aRow[2];
#pragma unroll
    for (int mt = 0; mt < 2; ++mt)
        aRow[mt] = hb + ((size_t)(m0 + mt * 16 + lrow) * kNL + n) * kD + lhi * 8;

    // 2-stage pipeline: regs hold slice k+1; LDS buf[p] holds slice k.
    float4 pg0 = *(const float4*)(gRow),  pg1 = *(const float4*)(gRow + 4);
    float4 pu0 = *(const float4*)(uRow),  pu1 = *(const float4*)(uRow + 4);
    *(v8bf*)(sG + sOff) = cvt8(pg0, pg1);
    *(v8bf*)(sU + sOff) = cvt8(pu0, pu1);
    pg0 = *(const float4*)(gRow + 32); pg1 = *(const float4*)(gRow + 36);
    pu0 = *(const float4*)(uRow + 32); pu1 = *(const float4*)(uRow + 36);
    __syncthreads();

    for (int kk = 0; kk < kD; kk += 32) {
        const int p  = (kk >> 5) & 1;
        const int q  = 1 - p;

        // stage slice k+1 into the other buffer (regs already resident)
        if (kk + 32 < kD) {
            *(v8bf*)(sG + q * SBUF + sOff) = cvt8(pg0, pg1);
            *(v8bf*)(sU + q * SBUF + sOff) = cvt8(pu0, pu1);
        }
        // issue global load for slice k+2 (latency covered by a full iteration)
        if (kk + 64 < kD) {
            pg0 = *(const float4*)(gRow + kk + 64); pg1 = *(const float4*)(gRow + kk + 68);
            pu0 = *(const float4*)(uRow + kk + 64); pu1 = *(const float4*)(uRow + kk + 68);
        }

        v16bf a[2];
#pragma unroll
        for (int mt = 0; mt < 2; ++mt) a[mt] = glb_a(aRow[mt] + kk);

        const __bf16* bG = sG + p * SBUF;
        const __bf16* bU = sU + p * SBUF;
#pragma unroll
        for (int t = 0; t < 4; ++t) {
            const int boff = (t * 16 + lrow) * SSTR + lhi * 8;
            v16bf bg = lds_b(bG + boff);
            v16bf bu = lds_b(bU + boff);
#pragma unroll
            for (int mt = 0; mt < 2; ++mt) {
                accG[mt][t] = wmma_bf16(a[mt], bg, accG[mt][t]);
                accU[mt][t] = wmma_bf16(a[mt], bu, accU[mt][t]);
            }
        }
        __syncthreads();   // single barrier per K-step
    }

    // epilogue: silu(gate)*up -> bf16 act, layout (B, N, I); fast v_rcp sigmoid
#pragma unroll
    for (int mt = 0; mt < 2; ++mt) {
#pragma unroll
        for (int t = 0; t < 4; ++t) {
            const int icol = i0 + t * 16 + lrow;
#pragma unroll
            for (int r = 0; r < 8; ++r) {
                const float g = accG[mt][t][r];
                const float u = accU[mt][t][r];
                const float s = g * __builtin_amdgcn_rcpf(1.0f + __expf(-g));
                const int brow = m0 + mt * 16 + lhi * 8 + r;
                act[((size_t)brow * kNL + n) * kI + icol] = (__bf16)(s * u);
            }
        }
    }
}

// ---------------------------------------------------------------------------
// Kernel 2: out = act · Wd^T  (act bf16 from ws, Wd f32 -> bf16 in LDS)
// grid: (kD/64, kNL); block: 256
// ---------------------------------------------------------------------------
__global__ __launch_bounds__(256, 1)
void down_kernel(const __bf16* __restrict__ act,
                 const float*  __restrict__ Wd,
                 float*        __restrict__ out)
{
    const int n   = blockIdx.y;
    const int d0  = blockIdx.x * 64;
    const int tid = threadIdx.x;
    const int lane = tid & 31;
    const int wave = tid >> 5;
    const int lrow = lane & 15;
    const int lhi  = lane >> 4;

    __shared__ __bf16 sD[2 * SBUF];

    v8f acc[2][4];
#pragma unroll
    for (int mt = 0; mt < 2; ++mt)
#pragma unroll
        for (int t = 0; t < 4; ++t) acc[mt][t] = {};

    const int wrow = tid >> 2;
    const int wk   = (tid & 3) * 8;
    const float* dRow = Wd + ((size_t)n * kD + (d0 + wrow)) * kI + wk;
    const int sOff = wrow * SSTR + wk;

    const int m0 = wave * 32;
    const __bf16* aRow[2];
#pragma unroll
    for (int mt = 0; mt < 2; ++mt)
        aRow[mt] = act + ((size_t)(m0 + mt * 16 + lrow) * kNL + n) * kI + lhi * 8;

    float4 pd0 = *(const float4*)(dRow), pd1 = *(const float4*)(dRow + 4);
    *(v8bf*)(sD + sOff) = cvt8(pd0, pd1);
    pd0 = *(const float4*)(dRow + 32); pd1 = *(const float4*)(dRow + 36);
    __syncthreads();

    for (int kk = 0; kk < kI; kk += 32) {
        const int p = (kk >> 5) & 1;
        const int q = 1 - p;

        if (kk + 32 < kI)
            *(v8bf*)(sD + q * SBUF + sOff) = cvt8(pd0, pd1);
        if (kk + 64 < kI) {
            pd0 = *(const float4*)(dRow + kk + 64);
            pd1 = *(const float4*)(dRow + kk + 68);
        }

        v16bf a[2];
#pragma unroll
        for (int mt = 0; mt < 2; ++mt) a[mt] = glb_a(aRow[mt] + kk);

        const __bf16* bD = sD + p * SBUF;
#pragma unroll
        for (int t = 0; t < 4; ++t) {
            const int boff = (t * 16 + lrow) * SSTR + lhi * 8;
            v16bf b = lds_b(bD + boff);
#pragma unroll
            for (int mt = 0; mt < 2; ++mt)
                acc[mt][t] = wmma_bf16(a[mt], b, acc[mt][t]);
        }
        __syncthreads();   // single barrier per K-step
    }

    // epilogue: f32 out, layout (B, N, D)
#pragma unroll
    for (int mt = 0; mt < 2; ++mt) {
#pragma unroll
        for (int t = 0; t < 4; ++t) {
            const int dcol = d0 + t * 16 + lrow;
#pragma unroll
            for (int r = 0; r < 8; ++r) {
                const int brow = m0 + mt * 16 + lhi * 8 + r;
                out[((size_t)brow * kNL + n) * kD + dcol] = acc[mt][t][r];
            }
        }
    }
}

// ---------------------------------------------------------------------------
extern "C" void kernel_launch(void* const* d_in, const int* in_sizes, int n_in,
                              void* d_out, int out_size, void* d_ws, size_t ws_size,
                              hipStream_t stream)
{
    (void)in_sizes; (void)n_in; (void)out_size; (void)ws_size;
    const float* h  = (const float*)d_in[0];
    const float* Wg = (const float*)d_in[1];
    const float* Wu = (const float*)d_in[2];
    const float* Wd = (const float*)d_in[3];
    float*  out = (float*)d_out;
    __bf16* act = (__bf16*)d_ws;                              // (B, N, I) bf16
    __bf16* hb  = (__bf16*)((char*)d_ws + kActBytes);         // (B, N, D) bf16

    dim3 blk(256);

    // h f32 -> bf16 (8,388,608 elems / 8 per thread)
    cvt_h_kernel<<<dim3((kB * kNL * kD) / (256 * 8)), blk, 0, stream>>>(h, hb);

    dim3 g1(kI / 64, kNL);   // 44 x 32
    gateup_silu_kernel<<<g1, blk, 0, stream>>>(hb, Wg, Wu, act);

    dim3 g2(kD / 64, kNL);   // 16 x 32
    down_kernel<<<g2, blk, 0, stream>>>(act, Wd, out);
}
